// Net_68049461838496
// MI455X (gfx1250) — compile-verified, matrix operand order checked by reading, not run on hardware
//
#include <hip/hip_runtime.h>
#include <hip/hip_bf16.h>

// PointNet++ segmentation forward for gfx1250 (MI455X).
// All GEMM work runs through v_wmma_f32_16x16x32_f16 with branch-free,
// 128-bit fragment loads (K-strides padded to multiples of 32 halves).
// Irregular work (FPS / radius top-K / kNN interp) uses LDS reductions.
// Peak workspace ~96MB -> stays resident in the 192MB L2.

typedef _Float16 half_t;
typedef __attribute__((ext_vector_type(8)))  _Float16 v8h;
typedef __attribute__((ext_vector_type(16))) _Float16 v16h;
typedef __attribute__((ext_vector_type(8)))  float    v8f;

#define B_      4
#define NPER_   4096
#define N0_     (B_ * NPER_)
#define FEAT_   4
#define N1_     820      // ceil(0.2*4096) per cloud
#define N2_     205      // ceil(0.25*820) per cloud
#define KNBR_   64

// ---------------------------------------------------------------------------
// Tiled WMMA GEMM: Y = act( X[MxK] @ W + b ), X row stride ldx (=Kpad, %32==0),
// weights pre-transposed+padded as Wt[N16pad x Kpad] (zero filled).
// One 16x16 tile per wave, 8 waves per block. Inner loop is branch-free:
// per fragment two 16B contiguous v8h loads (-> global_load_b128).
// ---------------------------------------------------------------------------
__global__ __launch_bounds__(256) void gemm_wmma(
    const half_t* __restrict__ X, int ldx,
    const half_t* __restrict__ Wt,
    const float* __restrict__ bias,
    half_t* __restrict__ Yh, float* __restrict__ Yf, int ldy,
    int M, int N, int KP, int relu)
{
    const int wave = threadIdx.x >> 5;
    const int lane = threadIdx.x & 31;
    const int tilesN = (N + 15) >> 4;
    const int tilesM = (M + 15) >> 4;
    const int tile = blockIdx.x * 8 + wave;
    if (tile >= tilesM * tilesN) return;
    const int tm = tile / tilesN;
    const int tn = tile - tm * tilesN;
    const int grp  = lane >> 4;          // lane group 0/1
    const int l16  = lane & 15;
    const int mrow = tm * 16 + l16;      // A row this lane feeds (< Mpad rows alloc'd)
    const int ncol = tn * 16 + l16;      // B column / C column this lane feeds

    // Per-lane fragment base: halves [8*grp, 8*grp+8) and [8*grp+16, 8*grp+24)
    const half_t* xp = X  + (size_t)mrow * ldx + 8 * grp;
    const half_t* wp = Wt + (size_t)ncol * KP  + 8 * grp;

    v8f acc = {};
    const int kTiles = KP >> 5;
    for (int kt = 0; kt < kTiles; ++kt) {
        v8h alo = *(const v8h*)(xp);
        v8h ahi = *(const v8h*)(xp + 16);
        v8h blo = *(const v8h*)(wp);
        v8h bhi = *(const v8h*)(wp + 16);
        v16h a = __builtin_shufflevector(alo, ahi, 0,1,2,3,4,5,6,7,8,9,10,11,12,13,14,15);
        v16h b = __builtin_shufflevector(blo, bhi, 0,1,2,3,4,5,6,7,8,9,10,11,12,13,14,15);
        acc = __builtin_amdgcn_wmma_f32_16x16x32_f16(
                  false, a, false, b, (short)0, acc, false, false);
        xp += 32;
        wp += 32;
    }

    const float bv = (bias && ncol < N) ? bias[ncol] : 0.f;
#pragma unroll
    for (int r = 0; r < 8; ++r) {
        const int m = tm * 16 + r + 8 * grp;     // C layout: vgpr r -> row r / r+8
        float v = acc[r] + bv;
        if (relu) v = v > 0.f ? v : 0.f;
        if (m < M && ncol < N) {
            if (Yh) Yh[(size_t)m * ldy + ncol] = (half_t)v;
            if (Yf) Yf[(size_t)m * ldy + ncol] = v;
        }
    }
}

// ---------------------------------------------------------------------------
// Farthest point sampling: one block per cloud, LDS-resident min-distance.
// ---------------------------------------------------------------------------
__global__ __launch_bounds__(256) void fps_kernel(
    const float* __restrict__ pos, int stride, int npts, int nsamp,
    int* __restrict__ outIdx)
{
    __shared__ float dd[4096];
    __shared__ float rv[256];
    __shared__ int   ri[256];
    __shared__ float cur[3];
    const int b = blockIdx.x, t = threadIdx.x;
    const int base = b * stride;
    if (t == 0) {
        cur[0] = pos[(size_t)base * 3 + 0];
        cur[1] = pos[(size_t)base * 3 + 1];
        cur[2] = pos[(size_t)base * 3 + 2];
        outIdx[b * nsamp] = base;
    }
    __syncthreads();
    for (int i = t; i < npts; i += 256) {
        float dx = pos[(size_t)(base + i) * 3 + 0] - cur[0];
        float dy = pos[(size_t)(base + i) * 3 + 1] - cur[1];
        float dz = pos[(size_t)(base + i) * 3 + 2] - cur[2];
        dd[i] = dx * dx + dy * dy + dz * dz;
    }
    __syncthreads();
    for (int s = 1; s < nsamp; ++s) {
        float bv = -1.f; int bi = 0;
        for (int i = t; i < npts; i += 256) {
            float v = dd[i];
            if (v > bv) { bv = v; bi = i; }
        }
        rv[t] = bv; ri[t] = bi;
        __syncthreads();
        for (int off = 128; off > 0; off >>= 1) {
            if (t < off) {
                if (rv[t + off] > rv[t] ||
                    (rv[t + off] == rv[t] && ri[t + off] < ri[t])) {
                    rv[t] = rv[t + off]; ri[t] = ri[t + off];
                }
            }
            __syncthreads();
        }
        if (t == 0) {
            const int g = base + ri[0];
            outIdx[b * nsamp + s] = g;
            cur[0] = pos[(size_t)g * 3 + 0];
            cur[1] = pos[(size_t)g * 3 + 1];
            cur[2] = pos[(size_t)g * 3 + 2];
        }
        __syncthreads();
        for (int i = t; i < npts; i += 256) {
            float dx = pos[(size_t)(base + i) * 3 + 0] - cur[0];
            float dy = pos[(size_t)(base + i) * 3 + 1] - cur[1];
            float dz = pos[(size_t)(base + i) * 3 + 2] - cur[2];
            float v = dx * dx + dy * dy + dz * dz;
            if (v < dd[i]) dd[i] = v;
        }
        __syncthreads();
    }
}

// ---------------------------------------------------------------------------
// K nearest within batch (+ radius validity): one block per query.
// ---------------------------------------------------------------------------
__global__ __launch_bounds__(256) void topk_kernel(
    const float* __restrict__ posq, const float* __restrict__ poss,
    int q_per_batch, int s_per_batch, int s_stride, int K, float r2,
    int* __restrict__ nbr, int* __restrict__ valid)
{
    __shared__ float d2[4096];
    __shared__ float rv[256];
    __shared__ int   ri[256];
    const int q = blockIdx.x, t = threadIdx.x;
    const int sbase = (q / q_per_batch) * s_stride;
    const float qx = posq[(size_t)q * 3 + 0];
    const float qy = posq[(size_t)q * 3 + 1];
    const float qz = posq[(size_t)q * 3 + 2];
    for (int i = t; i < s_per_batch; i += 256) {
        float dx = poss[(size_t)(sbase + i) * 3 + 0] - qx;
        float dy = poss[(size_t)(sbase + i) * 3 + 1] - qy;
        float dz = poss[(size_t)(sbase + i) * 3 + 2] - qz;
        d2[i] = dx * dx + dy * dy + dz * dz;
    }
    __syncthreads();
    for (int s = 0; s < K; ++s) {
        float bv = 3.0e38f; int bi = 0;
        for (int i = t; i < s_per_batch; i += 256) {
            float v = d2[i];
            if (v < bv) { bv = v; bi = i; }
        }
        rv[t] = bv; ri[t] = bi;
        __syncthreads();
        for (int off = 128; off > 0; off >>= 1) {
            if (t < off) {
                if (rv[t + off] < rv[t] ||
                    (rv[t + off] == rv[t] && ri[t + off] < ri[t])) {
                    rv[t] = rv[t + off]; ri[t] = ri[t + off];
                }
            }
            __syncthreads();
        }
        if (t == 0) {
            nbr[q * K + s] = sbase + ri[0];
            valid[q * K + s] = (rv[0] <= r2) ? 1 : 0;
            d2[ri[0]] = 3.0e38f;
        }
        __syncthreads();
    }
}

// ---------------------------------------------------------------------------
// kNN (k<=3) inverse-d2-weighted interpolation: one block per dest point.
// ---------------------------------------------------------------------------
__global__ __launch_bounds__(256) void interp_kernel(
    const float* __restrict__ posd, int d_per_batch,
    const float* __restrict__ poss, int s_per_batch, int s_stride,
    const half_t* __restrict__ xs, int C, int k,
    half_t* __restrict__ y, int ldy, int col0)
{
    __shared__ float d2[1024];
    __shared__ float rv[256];
    __shared__ int   ri[256];
    __shared__ int   nidx[3];
    __shared__ float nw[3];
    const int d = blockIdx.x, t = threadIdx.x;
    const int sbase = (d / d_per_batch) * s_stride;
    const float qx = posd[(size_t)d * 3 + 0];
    const float qy = posd[(size_t)d * 3 + 1];
    const float qz = posd[(size_t)d * 3 + 2];
    for (int i = t; i < s_per_batch; i += 256) {
        float dx = poss[(size_t)(sbase + i) * 3 + 0] - qx;
        float dy = poss[(size_t)(sbase + i) * 3 + 1] - qy;
        float dz = poss[(size_t)(sbase + i) * 3 + 2] - qz;
        d2[i] = dx * dx + dy * dy + dz * dz;
    }
    __syncthreads();
    for (int s = 0; s < k; ++s) {
        float bv = 3.0e38f; int bi = 0;
        for (int i = t; i < s_per_batch; i += 256) {
            float v = d2[i];
            if (v < bv) { bv = v; bi = i; }
        }
        rv[t] = bv; ri[t] = bi;
        __syncthreads();
        for (int off = 128; off > 0; off >>= 1) {
            if (t < off) {
                if (rv[t + off] < rv[t] ||
                    (rv[t + off] == rv[t] && ri[t + off] < ri[t])) {
                    rv[t] = rv[t + off]; ri[t] = ri[t + off];
                }
            }
            __syncthreads();
        }
        if (t == 0) {
            nidx[s] = sbase + ri[0];
            float dmin = rv[0] > 1e-16f ? rv[0] : 1e-16f;
            nw[s] = 1.f / dmin;
            d2[ri[0]] = 3.0e38f;
        }
        __syncthreads();
    }
    float wsum = 0.f;
    for (int s = 0; s < k; ++s) wsum += nw[s];
    if (wsum < 1e-16f) wsum = 1e-16f;
    for (int c = t; c < C; c += 256) {
        float acc = 0.f;
        for (int s = 0; s < k; ++s)
            acc += nw[s] * (float)xs[(size_t)nidx[s] * C + c];
        y[(size_t)d * ldy + col0 + c] = (half_t)(acc / wsum);
    }
}

// ---------------------------------------------------------------------------
// PointNetConv message build: row (q,j) = [x_src[nbr], pos_src[nbr]-pos_q[q]],
// zero-fills pad columns [F+3, Cpad).
// ---------------------------------------------------------------------------
__global__ void msg_build(
    const half_t* __restrict__ xsrc, int F,
    const float* __restrict__ possrc, const float* __restrict__ posq,
    const int* __restrict__ nbr, int K, int Mq, int Cpad,
    half_t* __restrict__ out)
{
    const int t = blockIdx.x * blockDim.x + threadIdx.x;
    if (t >= Mq * K) return;
    const int q = t / K;
    const int g = nbr[t];
    half_t* row = out + (size_t)t * Cpad;
    for (int f = 0; f < F; ++f) row[f] = xsrc[(size_t)g * F + f];
    row[F + 0] = (half_t)(possrc[(size_t)g * 3 + 0] - posq[(size_t)q * 3 + 0]);
    row[F + 1] = (half_t)(possrc[(size_t)g * 3 + 1] - posq[(size_t)q * 3 + 1]);
    row[F + 2] = (half_t)(possrc[(size_t)g * 3 + 2] - posq[(size_t)q * 3 + 2]);
    for (int c = F + 3; c < Cpad; ++c) row[c] = (half_t)0.f;
}

// Max aggregation over K neighbors with validity mask.
__global__ void max_agg(
    const half_t* __restrict__ H, const int* __restrict__ valid,
    int K, int M, int C, half_t* __restrict__ out)
{
    const int t = blockIdx.x * blockDim.x + threadIdx.x;
    if (t >= M * C) return;
    const int q = t / C, c = t - q * C;
    float best = -__builtin_inff();
    const half_t* hp = H + (size_t)q * K * C + c;
    const int* vp = valid + (size_t)q * K;
    for (int j = 0; j < K; ++j)
        if (vp[j]) {
            float v = (float)hp[(size_t)j * C];
            if (v > best) best = v;
        }
    out[t] = (half_t)best;
}

// Segment-max over contiguous per-batch rows.
__global__ void seg_max(
    const half_t* __restrict__ H, int rows, int C, half_t* __restrict__ out)
{
    const int t = blockIdx.x * blockDim.x + threadIdx.x;
    if (t >= B_ * C) return;
    const int b = t / C, c = t - b * C;
    float best = -__builtin_inff();
    for (int r = 0; r < rows; ++r) {
        float v = (float)H[((size_t)b * rows + r) * C + c];
        if (v > best) best = v;
    }
    out[t] = (half_t)best;
}

// Weight transpose+pad: Wt[NP x KP], zero outside [N x K].
__global__ void wt_conv(const float* __restrict__ W, half_t* __restrict__ Wt,
                        int K, int N, int KP, int NP)
{
    const int t = blockIdx.x * blockDim.x + threadIdx.x;
    if (t >= KP * NP) return;
    const int n = t / KP, k = t - n * KP;
    Wt[(size_t)n * KP + k] =
        (k < K && n < N) ? (half_t)W[(size_t)k * N + n] : (half_t)0.f;
}

__global__ void copy_f32_to_f16(const float* __restrict__ src, int M, int Cs,
                                half_t* __restrict__ dst, int ld, int col0)
{
    const int t = blockIdx.x * blockDim.x + threadIdx.x;
    if (t >= M * Cs) return;
    const int i = t / Cs, c = t - i * Cs;
    dst[(size_t)i * ld + col0 + c] = (half_t)src[(size_t)i * Cs + c];
}

__global__ void copy_f16(const half_t* __restrict__ src, int M, int Cs,
                         half_t* __restrict__ dst, int ld, int col0)
{
    const int t = blockIdx.x * blockDim.x + threadIdx.x;
    if (t >= M * Cs) return;
    const int i = t / Cs, c = t - i * Cs;
    dst[(size_t)i * ld + col0 + c] = src[(size_t)i * Cs + c];
}

__global__ void fill_f16(half_t* __restrict__ dst, int M, int ld, int col0,
                         int ncols)
{
    const int t = blockIdx.x * blockDim.x + threadIdx.x;
    if (t >= M * ncols) return;
    const int i = t / ncols, c = t - i * ncols;
    dst[(size_t)i * ld + col0 + c] = (half_t)0.f;
}

__global__ void gather_rows3(const float* __restrict__ src,
                             const int* __restrict__ idx, int M,
                             float* __restrict__ dst)
{
    const int t = blockIdx.x * blockDim.x + threadIdx.x;
    if (t >= M * 3) return;
    const int i = t / 3, c = t - i * 3;
    dst[(size_t)i * 3 + c] = src[(size_t)idx[i] * 3 + c];
}

// Per-batch broadcast of x3 (kNN with k=1 to a single centroid == gather).
__global__ void bcast_batch(const half_t* __restrict__ src, int rows_per_b,
                            int C, int M, half_t* __restrict__ dst, int ld,
                            int col0)
{
    const int t = blockIdx.x * blockDim.x + threadIdx.x;
    if (t >= M * C) return;
    const int i = t / C, c = t - i * C;
    dst[(size_t)i * ld + col0 + c] = src[(size_t)(i / rows_per_b) * C + c];
}

// ---------------------------------------------------------------------------
extern "C" void kernel_launch(void* const* d_in, const int* in_sizes, int n_in,
                              void* d_out, int out_size, void* d_ws, size_t ws_size,
                              hipStream_t stream)
{
    (void)in_sizes; (void)n_in; (void)out_size; (void)ws_size;
    const float* x   = (const float*)d_in[0];
    const float* pos = (const float*)d_in[1];
    float* outp = (float*)d_out;

    // Layers in pipeline order: sa1(3) sa2(3) sa3(3) fp1(2) fp2(2) fp3(3)
    // lin(2) out(1). d_in indices follow sorted-key pytree flatten.
    const int wIdx[19] = {23,25,27, 29,31,33, 35,37,39, 3,5, 7,9, 11,13,15, 17,19, 21};
    const int wK[19]   = {7,64,64, 131,128,128, 259,256,512, 1280,256, 384,256, 132,128,128, 128,128, 128};
    const int wN[19]   = {64,64,128, 128,128,256, 256,512,1024, 256,256, 256,128, 128,128,128, 128,128, 3};
    int wKP[19], wNP[19];
    for (int l = 0; l < 19; ++l) {
        wKP[l] = (wK[l] + 31) & ~31;   // pad K to multiple of 32 halves
        wNP[l] = (wN[l] + 15) & ~15;   // pad N to multiple of 16
    }

    // Workspace layout ------------------------------------------------------
    char* wsb = (char*)d_ws;
    size_t off = 0;
    auto alloc = [&](size_t bytes) -> void* {
        void* p = wsb + off;
        off = (off + bytes + 255) & ~(size_t)255;
        return p;
    };
    half_t* wt[19];
    for (int l = 0; l < 19; ++l)
        wt[l] = (half_t*)alloc((size_t)wKP[l] * wNP[l] * sizeof(half_t));
    half_t* xh   = (half_t*)alloc((size_t)N0_ * FEAT_ * sizeof(half_t));
    int*    idx1 = (int*)   alloc((size_t)B_ * N1_ * sizeof(int));
    float*  pos1 = (float*) alloc((size_t)B_ * N1_ * 3 * sizeof(float));
    int*    nbr1 = (int*)   alloc((size_t)B_ * N1_ * KNBR_ * sizeof(int));
    int*    val1 = (int*)   alloc((size_t)B_ * N1_ * KNBR_ * sizeof(int));
    half_t* x1h  = (half_t*)alloc((size_t)B_ * N1_ * 128 * sizeof(half_t));
    int*    idx2 = (int*)   alloc((size_t)B_ * N2_ * sizeof(int));
    float*  pos2 = (float*) alloc((size_t)B_ * N2_ * 3 * sizeof(float));
    int*    nbr2 = (int*)   alloc((size_t)B_ * N2_ * KNBR_ * sizeof(int));
    int*    val2 = (int*)   alloc((size_t)B_ * N2_ * KNBR_ * sizeof(int));
    half_t* x2h  = (half_t*)alloc((size_t)B_ * N2_ * 256 * sizeof(half_t));
    half_t* x3h  = (half_t*)alloc((size_t)B_ * 1024 * sizeof(half_t));
    half_t* f2h  = (half_t*)alloc((size_t)B_ * N2_ * 256 * sizeof(half_t));
    half_t* f1h  = (half_t*)alloc((size_t)B_ * N1_ * 128 * sizeof(half_t));
    half_t* f0h  = (half_t*)alloc((size_t)N0_ * 128 * sizeof(half_t));
    const size_t rowsSA1 = (size_t)B_ * N1_ * KNBR_;       // 209,920 (%16==0)
    half_t* bufA = (half_t*)alloc(rowsSA1 * 64  * sizeof(half_t)); // 26.9 MB
    half_t* bufB = (half_t*)alloc(rowsSA1 * 128 * sizeof(half_t)); // 53.7 MB

    auto g1 = [](size_t n) { return dim3((unsigned)((n + 255) / 256)); };
    auto gemm = [&](const half_t* X, int ldx, int l, half_t* Yh, float* Yf,
                    int ldy, int M, int relu) {
        const int tiles = ((M + 15) / 16) * ((wN[l] + 15) / 16);
        gemm_wmma<<<dim3((tiles + 7) / 8), dim3(256), 0, stream>>>(
            X, ldx, wt[l], (const float*)d_in[wIdx[l] + 1], Yh, Yf, ldy,
            M, wN[l], wKP[l], relu);
    };

    // Weight transpose + f16 conversion + zero pad --------------------------
    for (int l = 0; l < 19; ++l) {
        const size_t n = (size_t)wKP[l] * wNP[l];
        wt_conv<<<g1(n), 256, 0, stream>>>((const float*)d_in[wIdx[l]], wt[l],
                                           wK[l], wN[l], wKP[l], wNP[l]);
    }
    copy_f32_to_f16<<<g1((size_t)N0_ * FEAT_), 256, 0, stream>>>(
        x, N0_, FEAT_, xh, FEAT_, 0);

    const int M1 = B_ * N1_;   // 3280 (%16==0)
    const int M2 = B_ * N2_;   // 820  (tiles read to row 831; ws has room)

    // ---- SA1: fps -> radius top-64 -> msg[*,32] -> MLP 7-64-64-128 -> max
    fps_kernel<<<B_, 256, 0, stream>>>(pos, NPER_, NPER_, N1_, idx1);
    gather_rows3<<<g1((size_t)M1 * 3), 256, 0, stream>>>(pos, idx1, M1, pos1);
    topk_kernel<<<M1, 256, 0, stream>>>(pos1, pos, N1_, NPER_, NPER_, KNBR_,
                                        0.2f * 0.2f, nbr1, val1);
    msg_build<<<g1(rowsSA1), 256, 0, stream>>>(xh, FEAT_, pos, pos1, nbr1,
                                               KNBR_, M1, 32, bufA);
    gemm(bufA, 32, 0, bufB, nullptr, 64, (int)rowsSA1, 1);   // 7(32) -> 64
    gemm(bufB, 64, 1, bufA, nullptr, 64, (int)rowsSA1, 1);   // 64    -> 64
    gemm(bufA, 64, 2, bufB, nullptr, 128, (int)rowsSA1, 1);  // 64    -> 128
    max_agg<<<g1((size_t)M1 * 128), 256, 0, stream>>>(bufB, val1, KNBR_, M1,
                                                      128, x1h);

    // ---- SA2: msg[*,160] -> MLP 131-128-128-256 -> max
    fps_kernel<<<B_, 256, 0, stream>>>(pos1, N1_, N1_, N2_, idx2);
    gather_rows3<<<g1((size_t)M2 * 3), 256, 0, stream>>>(pos1, idx2, M2, pos2);
    topk_kernel<<<M2, 256, 0, stream>>>(pos2, pos1, N2_, N1_, N1_, KNBR_,
                                        0.4f * 0.4f, nbr2, val2);
    const size_t rowsSA2 = (size_t)M2 * KNBR_;               // 52,480 (%16==0)
    msg_build<<<g1(rowsSA2), 256, 0, stream>>>(x1h, 128, pos1, pos2, nbr2,
                                               KNBR_, M2, 160, bufA);
    gemm(bufA, 160, 3, bufB, nullptr, 128, (int)rowsSA2, 1); // 131(160) -> 128
    gemm(bufB, 128, 4, bufA, nullptr, 128, (int)rowsSA2, 1); // 128 -> 128
    gemm(bufA, 128, 5, bufB, nullptr, 256, (int)rowsSA2, 1); // 128 -> 256
    max_agg<<<g1((size_t)M2 * 256), 256, 0, stream>>>(bufB, val2, KNBR_, M2,
                                                      256, x2h);

    // ---- Global SA (sa3): concat[x2|pos2] pad->288 -> 256-512-1024 -> segmax
    copy_f16<<<g1((size_t)M2 * 256), 256, 0, stream>>>(x2h, M2, 256, bufA, 288, 0);
    copy_f32_to_f16<<<g1((size_t)M2 * 3), 256, 0, stream>>>(pos2, M2, 3, bufA,
                                                            288, 256);
    fill_f16<<<g1((size_t)832 * 29), 256, 0, stream>>>(bufA, 832, 288, 259, 29);
    gemm(bufA, 288, 6, bufB, nullptr, 256, M2, 1);           // 259(288) -> 256
    gemm(bufB, 256, 7, bufA, nullptr, 512, M2, 1);           // 256 -> 512
    gemm(bufA, 512, 8, bufB, nullptr, 1024, M2, 1);          // 512 -> 1024
    seg_max<<<g1((size_t)B_ * 1024), 256, 0, stream>>>(bufB, N2_, 1024, x3h);

    // ---- FP1: k=1 interp (per-batch broadcast) + concat -> 256 -> 256
    bcast_batch<<<g1((size_t)M2 * 1024), 256, 0, stream>>>(x3h, N2_, 1024, M2,
                                                           bufA, 1280, 0);
    copy_f16<<<g1((size_t)M2 * 256), 256, 0, stream>>>(x2h, M2, 256, bufA,
                                                       1280, 1024);
    gemm(bufA, 1280, 9, bufB, nullptr, 256, M2, 1);          // 1280 -> 256
    gemm(bufB, 256, 10, f2h, nullptr, 256, M2, 1);           // 256  -> 256

    // ---- FP2: k=3 interp pos2->pos1 + concat(384) -> 256 -> 128
    interp_kernel<<<M1, 256, 0, stream>>>(pos1, N1_, pos2, N2_, N2_, f2h, 256,
                                          3, bufA, 384, 0);
    copy_f16<<<g1((size_t)M1 * 128), 256, 0, stream>>>(x1h, M1, 128, bufA,
                                                       384, 256);
    gemm(bufA, 384, 11, bufB, nullptr, 256, M1, 1);          // 384 -> 256
    gemm(bufB, 256, 12, f1h, nullptr, 128, M1, 1);           // 256 -> 128

    // ---- FP3: k=3 interp pos1->pos + concat(132 pad->160) -> 128x3
    interp_kernel<<<N0_, 256, 0, stream>>>(pos, NPER_, pos1, N1_, N1_, f1h,
                                           128, 3, bufA, 160, 0);
    copy_f16<<<g1((size_t)N0_ * FEAT_), 256, 0, stream>>>(xh, N0_, FEAT_, bufA,
                                                          160, 128);
    fill_f16<<<g1((size_t)N0_ * 28), 256, 0, stream>>>(bufA, N0_, 160, 132, 28);
    gemm(bufA, 160, 13, bufB, nullptr, 128, N0_, 1);         // 132(160) -> 128
    gemm(bufB, 128, 14, bufA, nullptr, 128, N0_, 1);         // 128 -> 128
    gemm(bufA, 128, 15, f0h, nullptr, 128, N0_, 1);          // 128 -> 128

    // ---- Head: lin(128,128) x2 + out(128->3, f32, no relu)
    gemm(f0h, 128, 16, bufA, nullptr, 128, N0_, 1);          // 128 -> 128
    gemm(bufA, 128, 17, bufB, nullptr, 128, N0_, 1);         // 128 -> 128
    gemm(bufB, 128, 18, nullptr, outp, 3, N0_, 0);           // 128 -> 3
}
